// customGNN_27676769255977
// MI455X (gfx1250) — matrix operand order repaired and checked by reading, not compiled, not run on hardware
//
#include <hip/hip_runtime.h>
#include <hip/hip_bf16.h>

typedef __attribute__((ext_vector_type(16))) _Float16 v16h;
typedef __attribute__((ext_vector_type(8)))  _Float16 v8h;
typedef __attribute__((ext_vector_type(8)))  float    v8f;

#define NNODES 100000
#define NEDGES 1600000
#define HP     20      // padded node-feature row (19 used)
#define DPAD   64      // padded MLP-1 input (38 used), K for 2 chained WMMAs

// ---------------------------------------------------------------------------
// WMMA helper: D = A(16x32 f16) * B(32x16 f16) + C(16x16 f32)
// ---------------------------------------------------------------------------
static __device__ __forceinline__ v8f wmma_f16(v16h a, v16h b, v8f c) {
  return __builtin_amdgcn_wmma_f32_16x16x32_f16(false, a, false, b, (short)0, c,
                                                false, false);
}

static __device__ __forceinline__ v16h cat8(v8h lo, v8h hi) {
  return __builtin_shufflevector(lo, hi, 0,1,2,3,4,5,6,7,8,9,10,11,12,13,14,15);
}

// ---------------------------------------------------------------------------
// Pack fp32 weight matrices [K,16] into the wave32 WMMA B-operand layout:
//   element e of lane L (chunk c) holds W[(e<8 ? e : e+8) + 8*(L>=16) + 32*c][L&15]
// B64 pack = 2 chunks (K up to 64), B32 pack = 1 chunk (K up to 32).
// ---------------------------------------------------------------------------
__global__ void pack_weights_kernel(
    const float* __restrict__ eW1, const float* __restrict__ nW1,
    const float* __restrict__ oW1, const float* __restrict__ eW2,
    const float* __restrict__ nW2, const float* __restrict__ nW3,
    const float* __restrict__ oW2, _Float16* __restrict__ dst)
{
  const float* src; int K, off, chunks;
  switch (blockIdx.x) {
    case 0:  src = eW1;        K = 38; off = 0;    chunks = 2; break;
    case 1:  src = eW1 + 608;  K = 38; off = 1024; chunks = 2; break;
    case 2:  src = eW1 + 1216; K = 38; off = 2048; chunks = 2; break;
    case 3:  src = nW1;        K = 38; off = 3072; chunks = 2; break;
    case 4:  src = nW1 + 608;  K = 38; off = 4096; chunks = 2; break;
    case 5:  src = oW1;        K = 38; off = 5120; chunks = 2; break;
    case 6:  src = eW2;        K = 16; off = 6144; chunks = 1; break;
    case 7:  src = eW2 + 256;  K = 16; off = 6656; chunks = 1; break;
    case 8:  src = eW2 + 512;  K = 16; off = 7168; chunks = 1; break;
    case 9:  src = nW2;        K = 16; off = 7680; chunks = 1; break;
    case 10: src = nW2 + 256;  K = 16; off = 8192; chunks = 1; break;
    case 11: src = nW3;        K = 16; off = 8704; chunks = 1; break;
    case 12: src = nW3 + 256;  K = 16; off = 9216; chunks = 1; break;
    default: src = oW2;        K = 16; off = 9728; chunks = 1; break;
  }
  const int total = chunks * 512;
  for (int idx = threadIdx.x; idx < total; idx += blockDim.x) {
    int chunk = idx >> 9, rem = idx & 511, lane = rem >> 4, e = rem & 15;
    int k = (e < 8 ? e : e + 8) + ((lane >= 16) ? 8 : 0) + 32 * chunk;
    int n = lane & 15;
    float v = (k < K) ? src[k * 16 + n] : 0.0f;
    dst[off + idx] = (_Float16)v;
  }
}

// ---------------------------------------------------------------------------
// Node encoder: h[n] = [x@enc_W + enc_b (16), x (3), 0 pad]
// ---------------------------------------------------------------------------
__global__ void encoder_kernel(const float* __restrict__ x,
                               const float* __restrict__ W,
                               const float* __restrict__ b,
                               float* __restrict__ h)
{
  int n = blockIdx.x * blockDim.x + threadIdx.x;
  if (n >= NNODES) return;
  float x0 = x[n * 3 + 0], x1 = x[n * 3 + 1], x2 = x[n * 3 + 2];
  #pragma unroll
  for (int k = 0; k < 16; ++k)
    h[n * HP + k] = x0 * W[k] + x1 * W[16 + k] + x2 * W[32 + k] + b[k];
  h[n * HP + 16] = x0;
  h[n * HP + 17] = x1;
  h[n * HP + 18] = x2;
  h[n * HP + 19] = 0.0f;
}

__global__ void zero_kernel(float* __restrict__ p, int n)
{
  int i = blockIdx.x * blockDim.x + threadIdx.x;
  if (i < n) p[i] = 0.0f;
}

// Fetch one padded 20-float row (80 B, 16-B aligned) as 5x b128.
static __device__ __forceinline__ void load_row20(const float* __restrict__ src,
                                                  float* __restrict__ buf)
{
  const float4* s4 = (const float4*)src;
  float4 q0 = s4[0], q1 = s4[1], q2 = s4[2], q3 = s4[3], q4 = s4[4];
  *(float4*)&buf[0]  = q0;
  *(float4*)&buf[4]  = q1;
  *(float4*)&buf[8]  = q2;
  *(float4*)&buf[12] = q3;
  *(float4*)&buf[16] = q4;
}

// ---------------------------------------------------------------------------
// Edge kernel: one wave32 owns 16 edges.
//   L1: [16e x 38] @ W1(38x16)  -> 2 chained WMMAs (K padded to 64)
//   L2: [16e x 16] @ W2(16x16)  -> 1 WMMA via LDS transpose
//   L3: dot with W3(16) + b3, relu, sigmoid -> edge gate
//   scatter: msgs[start] += h[end] * gate (19 atomic f32 adds / edge)
// ---------------------------------------------------------------------------
__global__ __launch_bounds__(256) void edge_kernel(
    const float* __restrict__ h, const int* __restrict__ eidx,
    const _Float16* __restrict__ w1p, const _Float16* __restrict__ w2p,
    const float* __restrict__ b1, const float* __restrict__ b2,
    const float* __restrict__ W3, const float* __restrict__ b3,
    float* __restrict__ msgs, int numTiles)
{
  __shared__ _Float16 smem[8 * 1024];
  const int wave = threadIdx.x >> 5;
  const int lane = threadIdx.x & 31;
  const int tile = blockIdx.x * 8 + wave;
  if (tile >= numTiles) return;
  _Float16* fbuf = &smem[wave * 1024];

  const int eBase = tile * 16;
  const int j  = lane >> 1;      // edge slot 0..15
  const int hh = lane & 1;       // which half of the 38 features
  const int e  = eBase + j;
  const int s  = eidx[e];            // start (segment id)
  const int t  = eidx[NEDGES + e];   // end   (message source)

  // Stage ei = [h[start] (19) | h[end] (19) | zeros] as f16 rows of 64.
  // hh==0 lane stages h[start] into cols 0..18, hh==1 stages h[end] into 19..37.
  float buf[20];
  {
    load_row20(h + (hh == 0 ? s : t) * HP, buf);
    _Float16* row = fbuf + j * DPAD + hh * 19;
    #pragma unroll
    for (int k = 0; k < 19; ++k) row[k] = (_Float16)buf[k];
    _Float16* pad = fbuf + j * DPAD + 38 + hh * 13;
    #pragma unroll
    for (int k = 0; k < 13; ++k) pad[k] = (_Float16)0.0f;
  }
  asm volatile("s_wait_dscnt 0" ::: "memory");  // wave-internal LDS RAW

  const int m = lane & 15;   // A row / C column
  const int g = lane >> 4;   // lane half

  // Layer 1: two chained 16x16x32 WMMAs over K=0..63
  const _Float16* arow = fbuf + m * DPAD;
  v16h A0 = cat8(*(const v8h*)(arow + g * 8),      *(const v8h*)(arow + 16 + g * 8));
  v16h A1 = cat8(*(const v8h*)(arow + 32 + g * 8), *(const v8h*)(arow + 48 + g * 8));
  v16h B0 = *(const v16h*)(w1p + lane * 16);
  v16h B1 = *(const v16h*)(w1p + 512 + lane * 16);
  v8f c = {};
  c = wmma_f16(A0, B0, c);
  c = wmma_f16(A1, B1, c);

  // bias + relu, transpose C-layout -> A-layout via LDS (16x16, stride 32)
  const float bias1 = b1[m];
  #pragma unroll
  for (int r = 0; r < 8; ++r) {
    float v = fmaxf(c[r] + bias1, 0.0f);
    fbuf[(r + 8 * g) * 32 + m] = (_Float16)v;
  }
  asm volatile("s_wait_dscnt 0" ::: "memory");

  // Layer 2: one WMMA, K=16 used (upper K zero)
  const _Float16* arow2 = fbuf + m * 32;
  v8h zero8 = {};
  v16h A2 = cat8(*(const v8h*)(arow2 + g * 8), zero8);
  v16h B2 = *(const v16h*)(w2p + lane * 16);
  v8f c2 = {};
  c2 = wmma_f16(A2, B2, c2);

  // Layer 3: relu(c2+b2) . W3, butterfly-reduce across the 16-lane half
  const float bias2 = b2[m];
  const float w3v = W3[m];
  float acc[8];
  #pragma unroll
  for (int r = 0; r < 8; ++r) {
    float v = fmaxf(c2[r] + bias2, 0.0f) * w3v;
    v += __shfl_xor(v, 1, 32);
    v += __shfl_xor(v, 2, 32);
    v += __shfl_xor(v, 4, 32);
    v += __shfl_xor(v, 8, 32);
    acc[r] = v;  // all lanes in half g hold sums for edges 8g..8g+7
  }
  // this lane's edge j lives in its own half: pick register r = j&7
  float sum = acc[0];
  #pragma unroll
  for (int r = 1; r < 8; ++r) sum = ((j & 7) == r) ? acc[r] : sum;
  const float lin  = fmaxf(sum + b3[0], 0.0f);
  const float gate = 1.0f / (1.0f + __expf(-lin));

  // scatter: msgs[start] += h[end] * gate (split 19 dims across lane pair).
  // hh==1 lane already holds h[end] in buf; hh==0 reloads its 10 dims as b128.
  float* mrow = msgs + s * HP;
  if (hh == 0) {
    float tb[12];
    const float4* tv = (const float4*)(h + t * HP);
    float4 r0 = tv[0], r1 = tv[1], r2 = tv[2];
    *(float4*)&tb[0] = r0; *(float4*)&tb[4] = r1; *(float4*)&tb[8] = r2;
    #pragma unroll
    for (int k = 0; k < 10; ++k) atomicAdd(&mrow[k], tb[k] * gate);
  } else {
    #pragma unroll
    for (int k = 10; k < 19; ++k) atomicAdd(&mrow[k], buf[k] * gate);
  }
}

// ---------------------------------------------------------------------------
// Node kernel (iterations 0..GI-2): one wave owns 16 nodes.
//   ni = [h (19) | msgs (19)]; 3 WMMA layers; h_new = relu(L3)+x0, +x residual
// ---------------------------------------------------------------------------
__global__ __launch_bounds__(256) void node_kernel(
    float* __restrict__ h, const float* __restrict__ msgs,
    const float* __restrict__ x,
    const _Float16* __restrict__ w1p, const _Float16* __restrict__ w2p,
    const _Float16* __restrict__ w3p,
    const float* __restrict__ b1, const float* __restrict__ b2,
    const float* __restrict__ b3, int numTiles)
{
  __shared__ _Float16 smem[8 * 1024];
  const int wave = threadIdx.x >> 5;
  const int lane = threadIdx.x & 31;
  const int tile = blockIdx.x * 8 + wave;
  if (tile >= numTiles) return;
  _Float16* fbuf = &smem[wave * 1024];

  const int nBase = tile * 16;
  const int j  = lane >> 1;
  const int hh = lane & 1;
  const int node = nBase + j;

  {
    float buf[20];
    load_row20((hh == 0 ? h : msgs) + node * HP, buf);
    _Float16* row = fbuf + j * DPAD + hh * 19;
    #pragma unroll
    for (int k = 0; k < 19; ++k) row[k] = (_Float16)buf[k];
    _Float16* pad = fbuf + j * DPAD + 38 + hh * 13;
    #pragma unroll
    for (int k = 0; k < 13; ++k) pad[k] = (_Float16)0.0f;
  }
  asm volatile("s_wait_dscnt 0" ::: "memory");

  const int m = lane & 15;
  const int g = lane >> 4;

  const _Float16* arow = fbuf + m * DPAD;
  v16h A0 = cat8(*(const v8h*)(arow + g * 8),      *(const v8h*)(arow + 16 + g * 8));
  v16h A1 = cat8(*(const v8h*)(arow + 32 + g * 8), *(const v8h*)(arow + 48 + g * 8));
  v16h B0 = *(const v16h*)(w1p + lane * 16);
  v16h B1 = *(const v16h*)(w1p + 512 + lane * 16);
  v8f c = {};
  c = wmma_f16(A0, B0, c);
  c = wmma_f16(A1, B1, c);

  const float bias1 = b1[m];
  #pragma unroll
  for (int r = 0; r < 8; ++r) {
    float v = fmaxf(c[r] + bias1, 0.0f);
    fbuf[(r + 8 * g) * 32 + m] = (_Float16)v;
  }
  asm volatile("s_wait_dscnt 0" ::: "memory");

  v8h zero8 = {};
  const _Float16* arow2 = fbuf + m * 32;
  v16h A2 = cat8(*(const v8h*)(arow2 + g * 8), zero8);
  v16h B2 = *(const v16h*)(w2p + lane * 16);
  v8f c2 = {};
  c2 = wmma_f16(A2, B2, c2);

  const float bias2 = b2[m];
  #pragma unroll
  for (int r = 0; r < 8; ++r) {
    float v = fmaxf(c2[r] + bias2, 0.0f);
    fbuf[(r + 8 * g) * 32 + m] = (_Float16)v;
  }
  asm volatile("s_wait_dscnt 0" ::: "memory");

  const _Float16* arow3 = fbuf + m * 32;
  v16h A3 = cat8(*(const v8h*)(arow3 + g * 8), zero8);
  v16h B3 = *(const v16h*)(w3p + lane * 16);
  v8f c3 = {};
  c3 = wmma_f16(A3, B3, c3);

  // h_new[:,0:16] = relu(c3 + b3) + x0 ; h_new[:,16:19] = x0[:,16:19] + x
  const float bias3 = b3[m];
  #pragma unroll
  for (int r = 0; r < 8; ++r) {
    int nd = nBase + r + 8 * g;
    float v = fmaxf(c3[r] + bias3, 0.0f);
    h[nd * HP + m] = v + h[nd * HP + m];
  }
  if (hh == 0) {
    #pragma unroll
    for (int f = 0; f < 3; ++f)
      h[node * HP + 16 + f] += x[node * 3 + f];
  }
}

// ---------------------------------------------------------------------------
// Output kernel (last iteration): L1, L2 via WMMA; L3 dot + sigmoid -> d_out
// ---------------------------------------------------------------------------
__global__ __launch_bounds__(256) void out_kernel(
    const float* __restrict__ h, const float* __restrict__ msgs,
    const _Float16* __restrict__ w1p, const _Float16* __restrict__ w2p,
    const float* __restrict__ b1, const float* __restrict__ b2,
    const float* __restrict__ W3, const float* __restrict__ b3,
    float* __restrict__ out, int numTiles)
{
  __shared__ _Float16 smem[8 * 1024];
  const int wave = threadIdx.x >> 5;
  const int lane = threadIdx.x & 31;
  const int tile = blockIdx.x * 8 + wave;
  if (tile >= numTiles) return;
  _Float16* fbuf = &smem[wave * 1024];

  const int nBase = tile * 16;
  const int j  = lane >> 1;
  const int hh = lane & 1;
  const int node = nBase + j;

  {
    float buf[20];
    load_row20((hh == 0 ? h : msgs) + node * HP, buf);
    _Float16* row = fbuf + j * DPAD + hh * 19;
    #pragma unroll
    for (int k = 0; k < 19; ++k) row[k] = (_Float16)buf[k];
    _Float16* pad = fbuf + j * DPAD + 38 + hh * 13;
    #pragma unroll
    for (int k = 0; k < 13; ++k) pad[k] = (_Float16)0.0f;
  }
  asm volatile("s_wait_dscnt 0" ::: "memory");

  const int m = lane & 15;
  const int g = lane >> 4;

  const _Float16* arow = fbuf + m * DPAD;
  v16h A0 = cat8(*(const v8h*)(arow + g * 8),      *(const v8h*)(arow + 16 + g * 8));
  v16h A1 = cat8(*(const v8h*)(arow + 32 + g * 8), *(const v8h*)(arow + 48 + g * 8));
  v16h B0 = *(const v16h*)(w1p + lane * 16);
  v16h B1 = *(const v16h*)(w1p + 512 + lane * 16);
  v8f c = {};
  c = wmma_f16(A0, B0, c);
  c = wmma_f16(A1, B1, c);

  const float bias1 = b1[m];
  #pragma unroll
  for (int r = 0; r < 8; ++r) {
    float v = fmaxf(c[r] + bias1, 0.0f);
    fbuf[(r + 8 * g) * 32 + m] = (_Float16)v;
  }
  asm volatile("s_wait_dscnt 0" ::: "memory");

  v8h zero8 = {};
  const _Float16* arow2 = fbuf + m * 32;
  v16h A2 = cat8(*(const v8h*)(arow2 + g * 8), zero8);
  v16h B2 = *(const v16h*)(w2p + lane * 16);
  v8f c2 = {};
  c2 = wmma_f16(A2, B2, c2);

  const float bias2 = b2[m];
  const float w3v = W3[m];
  float acc[8];
  #pragma unroll
  for (int r = 0; r < 8; ++r) {
    float v = fmaxf(c2[r] + bias2, 0.0f) * w3v;
    v += __shfl_xor(v, 1, 32);
    v += __shfl_xor(v, 2, 32);
    v += __shfl_xor(v, 4, 32);
    v += __shfl_xor(v, 8, 32);
    acc[r] = v;
  }
  if (m == 0) {  // lanes 0 and 16 each write 8 node outputs
    #pragma unroll
    for (int r = 0; r < 8; ++r)
      out[nBase + r + 8 * g] = 1.0f / (1.0f + __expf(-(acc[r] + b3[0])));
  }
}

// ---------------------------------------------------------------------------
extern "C" void kernel_launch(void* const* d_in, const int* in_sizes, int n_in,
                              void* d_out, int out_size, void* d_ws, size_t ws_size,
                              hipStream_t stream)
{
  const float* x     = (const float*)d_in[0];
  const int*   eidx  = (const int*)  d_in[1];
  const float* enc_W = (const float*)d_in[2];
  const float* enc_b = (const float*)d_in[3];
  const float* eW1   = (const float*)d_in[4];
  const float* eb1   = (const float*)d_in[5];
  const float* eW2   = (const float*)d_in[6];
  const float* eb2   = (const float*)d_in[7];
  const float* eW3   = (const float*)d_in[8];
  const float* eb3   = (const float*)d_in[9];
  const float* nW1   = (const float*)d_in[10];
  const float* nb1   = (const float*)d_in[11];
  const float* nW2   = (const float*)d_in[12];
  const float* nb2   = (const float*)d_in[13];
  const float* nW3   = (const float*)d_in[14];
  const float* nb3   = (const float*)d_in[15];
  const float* oW1   = (const float*)d_in[16];
  const float* ob1   = (const float*)d_in[17];
  const float* oW2   = (const float*)d_in[18];
  const float* ob2   = (const float*)d_in[19];
  const float* oW3   = (const float*)d_in[20];
  const float* ob3   = (const float*)d_in[21];

  float* h    = (float*)d_ws;                               // N*20 f32 (8 MB)
  float* msgs = h + (size_t)NNODES * HP;                    // N*20 f32 (8 MB)
  _Float16* packs = (_Float16*)((char*)d_ws + (size_t)2 * NNODES * HP * sizeof(float));

  pack_weights_kernel<<<14, 256, 0, stream>>>(eW1, nW1, oW1, eW2, nW2, nW3, oW2, packs);
  encoder_kernel<<<(NNODES + 255) / 256, 256, 0, stream>>>(x, enc_W, enc_b, h);

  const int edgeTiles  = NEDGES / 16;             // 100000 (exact)
  const int nodeTiles  = NNODES / 16;             // 6250   (exact)
  const int edgeBlocks = (edgeTiles + 7) / 8;
  const int nodeBlocks = (nodeTiles + 7) / 8;

  for (int i = 0; i < 3; ++i) {
    zero_kernel<<<(NNODES * HP + 255) / 256, 256, 0, stream>>>(msgs, NNODES * HP);
    edge_kernel<<<edgeBlocks, 256, 0, stream>>>(
        h, eidx, packs + i * 1024, packs + 6144 + i * 512,
        eb1 + i * 16, eb2 + i * 16, eW3 + i * 16, eb3 + i, msgs, edgeTiles);
    if (i < 2) {
      node_kernel<<<nodeBlocks, 256, 0, stream>>>(
          h, msgs, x,
          packs + 3072 + i * 1024, packs + 7680 + i * 512, packs + 8704 + i * 512,
          nb1 + i * 16, nb2 + i * 16, nb3 + i * 16, nodeTiles);
    } else {
      out_kernel<<<nodeBlocks, 256, 0, stream>>>(
          h, msgs, packs + 5120, packs + 9728,
          ob1, ob2, oW3, ob3, (float*)d_out, nodeTiles);
    }
  }
  (void)in_sizes; (void)n_in; (void)out_size; (void)ws_size;
}
